// GTrans_C_89988154786409
// MI455X (gfx1250) — compile-verified, matrix operand order checked by reading, not run on hardware
//
#include <hip/hip_runtime.h>
#include <hip/hip_bf16.h>
#include <math.h>
#include <string.h>

#define H_   3
#define C_   64
#define HC_  192
#define B_   256
#define N0_  128
#define DEG_ 8
#define NT_  (B_ * N0_)          // 32768 nodes
#define E_   (B_ * N0_ * DEG_)   // 262144 edges
#define FIN_ 128
#define ED_  16
#define CDIV(a, b) (((a) + (b) - 1) / (b))

typedef __attribute__((ext_vector_type(16))) _Float16 v16h;
typedef __attribute__((ext_vector_type(8)))  _Float16 v8h;
typedef __attribute__((ext_vector_type(8)))  float    v8f;

// ---- monotonic float<->uint encoding for atomic segment-max ----------------
static __device__ __forceinline__ unsigned f2key(float f) {
  unsigned b = __float_as_uint(f);
  return (b & 0x80000000u) ? ~b : (b | 0x80000000u);
}
static __device__ __forceinline__ float key2f(unsigned k) {
  unsigned b = (k & 0x80000000u) ? (k & 0x7fffffffu) : ~k;
  return __uint_as_float(b);
}

// ---- utility fills ---------------------------------------------------------
__global__ void fill_f32(float* p, float v, int n) {
  int t = blockIdx.x * blockDim.x + threadIdx.x;
  if (t < n) p[t] = v;
}
__global__ void fill_u32(unsigned* p, unsigned v, int n) {
  int t = blockIdx.x * blockDim.x + threadIdx.x;
  if (t < n) p[t] = v;
}
__global__ void copy_edges(const int* ei, int* srcb, int* dstb, int* maskb) {
  int t = blockIdx.x * blockDim.x + threadIdx.x;
  if (t >= E_) return;
  srcb[t] = ei[t];
  dstb[t] = ei[E_ + t];
  maskb[t] = 1;
}

// ---- fp32 -> f16 activation convert (vector-4) -----------------------------
__global__ void cvt_f16(const float* __restrict__ a, _Float16* __restrict__ o, int n) {
  int t = (blockIdx.x * blockDim.x + threadIdx.x) * 4;
  if (t + 3 < n) {
    float4 v = *(const float4*)(a + t);
    o[t] = (_Float16)v.x; o[t + 1] = (_Float16)v.y;
    o[t + 2] = (_Float16)v.z; o[t + 3] = (_Float16)v.w;
  } else {
    for (int i = t; i < n; ++i) o[i] = (_Float16)a[i];
  }
}

// ---- pack W[K,N] fp32 into B-fragment layout: [tn][kblk][lane][16 halfs] ---
// lane<16: halves = K kb..kb+7 / kb+16..kb+23 ; lane>=16: kb+=8 (ISA layout)
__global__ void pack_b(const float* __restrict__ W, _Float16* __restrict__ Wp,
                       int K, int N) {
  int t = blockIdx.x * blockDim.x + threadIdx.x;
  int total = (N / 16) * (K / 32) * 32;
  if (t >= total) return;
  int lane = t & 31;
  int rest = t >> 5;
  int kblk = rest % (K / 32);
  int tn = rest / (K / 32);
  int hs = lane >> 4, l15 = lane & 15;
  int ncol = tn * 16 + l15;
  int kb = kblk * 32 + hs * 8;
  _Float16* o = Wp + (size_t)t * 16;
#pragma unroll
  for (int h = 0; h < 8; ++h) {
    o[h] = (_Float16)W[(kb + h) * N + ncol];
    o[h + 8] = (_Float16)W[(kb + 16 + h) * N + ncol];
  }
}

// ---- load one 16x32 A fragment (two contiguous 16B chunks) -----------------
static __device__ __forceinline__ v16h load_afrag(const _Float16* __restrict__ ap) {
  v8h lo = *(const v8h*)ap;          // K kb..kb+7   (16B)
  v8h hi = *(const v8h*)(ap + 16);   // K kb+16..+23 (16B)
  v16h a;
#pragma unroll
  for (int h = 0; h < 8; ++h) { a[h] = lo[h]; a[h + 8] = hi[h]; }
  return a;
}

// ---- WMMA GEMM: out[M,N] = act(A[M,K] @ W[K,N] + bias), opt BN -------------
// one wave32 per 64x16 tile block: 4 accumulators, 4 wmma per 32-K step.
// All 4 A fragments loaded into distinct registers before the WMMA burst so
// the loop body is: load clause -> one wait -> 4 back-to-back v_wmma.
// A: f16 row-major; Wp: pre-packed B fragments. K % 32 == 0, M % 64 == 0.
__global__ void gemm16(const _Float16* __restrict__ A, const _Float16* __restrict__ Wp,
                       const float* __restrict__ bias, float* __restrict__ out,
                       int M, int N, int K, int act,
                       const float* __restrict__ bng, const float* __restrict__ bnb) {
  const int lane = threadIdx.x & 31;
  const int tm = blockIdx.x;       // 64-row block
  const int tn = blockIdx.y;       // 16-col block
  const int hs = lane >> 4, l15 = lane & 15;
  const int ncol = tn * 16 + l15;
  const int kblocks = K >> 5;
  v8f acc0 = {}, acc1 = {}, acc2 = {}, acc3 = {};
  const _Float16* bp = Wp + ((size_t)tn * kblocks * 32 + lane) * 16;
  const _Float16* ap0 = A + (size_t)(tm * 64 + l15) * K + hs * 8;
  for (int kb = 0; kb < kblocks; ++kb) {
    v16h b = *(const v16h*)(bp + (size_t)kb * 512);
    const _Float16* ak = ap0 + kb * 32;
    v16h a0 = load_afrag(ak);
    v16h a1 = load_afrag(ak + (size_t)16 * K);
    v16h a2 = load_afrag(ak + (size_t)32 * K);
    v16h a3 = load_afrag(ak + (size_t)48 * K);
    acc0 = __builtin_amdgcn_wmma_f32_16x16x32_f16(false, a0, false, b, (short)0, acc0, false, false);
    acc1 = __builtin_amdgcn_wmma_f32_16x16x32_f16(false, a1, false, b, (short)0, acc1, false, false);
    acc2 = __builtin_amdgcn_wmma_f32_16x16x32_f16(false, a2, false, b, (short)0, acc2, false, false);
    acc3 = __builtin_amdgcn_wmma_f32_16x16x32_f16(false, a3, false, b, (short)0, acc3, false, false);
  }
  float bv = bias ? bias[ncol] : 0.f;
  float g = 1.f, bb = 0.f;
  if (bng) { g = bng[ncol] * rsqrtf(1.f + 1e-5f); bb = bnb[ncol]; }
#pragma unroll
  for (int i = 0; i < 4; ++i) {
    const v8f acc = (i == 0) ? acc0 : (i == 1) ? acc1 : (i == 2) ? acc2 : acc3;
    const int mb = tm * 64 + i * 16 + hs * 8;   // C/D layout: vgpr r -> row mb+r
#pragma unroll
    for (int r = 0; r < 8; ++r) {
      float x = acc[r] + bv;
      if (act) x = fmaxf(x, 0.f);
      if (bng) x = x * g + bb;
      out[(size_t)(mb + r) * N + ncol] = x;
    }
  }
}

// ---- attention pass 1: alpha + atomic segment max --------------------------
__global__ void attn_alpha(const float* __restrict__ q, const float* __restrict__ k,
                           const float* __restrict__ ea, const float* __restrict__ We,
                           const int* __restrict__ srcb, const int* __restrict__ dstb,
                           const int* __restrict__ maskb,
                           float* __restrict__ alpha, unsigned* __restrict__ amaxb) {
  int t = blockIdx.x * blockDim.x + threadIdx.x;
  if (t >= E_ * H_) return;
  int e = t / H_, h = t % H_;
  int s = srcb[e], d = dstb[e];
  float ew[ED_];
#pragma unroll
  for (int i = 0; i < ED_; ++i) ew[i] = ea[e * ED_ + i];
  float sum = 0.f;
  for (int c = 0; c < C_; ++c) {
    float ev = 0.f;
#pragma unroll
    for (int i = 0; i < ED_; ++i) ev += ew[i] * We[i * HC_ + h * C_ + c];
    sum += q[(size_t)d * HC_ + h * C_ + c] * (k[(size_t)s * HC_ + h * C_ + c] + ev);
  }
  sum *= 0.125f;                   // 1/sqrt(64)
  if (!maskb[e]) sum = -1e9f;
  alpha[t] = sum;
  atomicMax(&amaxb[d * H_ + h], f2key(sum));
}

// ---- attention pass 2: ex = exp(alpha - amax)*mask ; den += ex -------------
__global__ void attn_exden(float* __restrict__ alpha, const unsigned* __restrict__ amaxb,
                           const int* __restrict__ dstb, const int* __restrict__ maskb,
                           float* __restrict__ den) {
  int t = blockIdx.x * blockDim.x + threadIdx.x;
  if (t >= E_ * H_) return;
  int e = t / H_, h = t % H_;
  int d = dstb[e];
  float ex = 0.f;
  if (maskb[e]) ex = expf(alpha[t] - key2f(amaxb[d * H_ + h]));
  alpha[t] = ex;
  if (ex != 0.f) atomicAdd(&den[d * H_ + h], ex);
}

// ---- attention pass 3: out[dst] += (v[src]+e) * ex/den ---------------------
__global__ void attn_msg(const float* __restrict__ alpha, const float* __restrict__ den,
                         const float* __restrict__ v, const float* __restrict__ ea,
                         const float* __restrict__ We,
                         const int* __restrict__ srcb, const int* __restrict__ dstb,
                         const int* __restrict__ maskb, float* __restrict__ outa) {
  int t = blockIdx.x * blockDim.x + threadIdx.x;
  if (t >= E_ * H_) return;
  int e = t / H_, h = t % H_;
  if (!maskb[e]) return;
  int s = srcb[e], d = dstb[e];
  float w = alpha[t] / fmaxf(den[d * H_ + h], 1e-16f);
  if (w == 0.f) return;
  float ew[ED_];
#pragma unroll
  for (int i = 0; i < ED_; ++i) ew[i] = ea[e * ED_ + i];
  for (int c = 0; c < C_; ++c) {
    float ev = 0.f;
#pragma unroll
    for (int i = 0; i < ED_; ++i) ev += ew[i] * We[i * HC_ + h * C_ + c];
    atomicAdd(&outa[(size_t)d * HC_ + h * C_ + c],
              (v[(size_t)s * HC_ + h * C_ + c] + ev) * w);
  }
}

// ---- beta gate: x = sigmoid([out,xr,out-xr]@Wb)*xr + (1-b)*out -------------
__global__ void beta_combine(const float* __restrict__ outa, const float* __restrict__ xr,
                             const float* __restrict__ Wb, float* __restrict__ tc, int N) {
  int n = blockIdx.x * blockDim.x + threadIdx.x;
  if (n >= N) return;
  const float* o = outa + (size_t)n * HC_;
  const float* r = xr + (size_t)n * HC_;
  float z = 0.f;
  for (int j = 0; j < HC_; ++j)
    z += o[j] * Wb[j] + r[j] * Wb[HC_ + j] + (o[j] - r[j]) * Wb[2 * HC_ + j];
  float be = 1.f / (1.f + expf(-z));
  for (int j = 0; j < HC_; ++j)
    tc[(size_t)n * HC_ + j] = be * r[j] + (1.f - be) * o[j];
}

// ---- per-graph TopK pooling (ratio 0.5), rank-by-count selection -----------
__global__ void topk_pool(const float* __restrict__ x, const float* __restrict__ w,
                          int n, int kk, float* __restrict__ xo, int* __restrict__ newpos) {
  __shared__ float s[128];
  int g = blockIdx.x, t = threadIdx.x;
  float wn = 0.f;
  for (int f = 0; f < 64; ++f) wn += w[f] * w[f];
  wn = sqrtf(wn);
  int node = g * n + t;
  float sc = 0.f;
  if (t < n) {
    float d = 0.f;
    for (int f = 0; f < 64; ++f) d += x[(size_t)node * 64 + f] * w[f];
    sc = tanhf(d / wn);
    s[t] = sc;
  }
  __syncthreads();
  if (t >= n) return;
  int rank = 0;
  for (int j = 0; j < n; ++j) {
    float sj = s[j];
    rank += (sj > sc) || (sj == sc && j < t);
  }
  if (rank < kk) {
    int pos = g * kk + rank;
    for (int f = 0; f < 64; ++f) xo[(size_t)pos * 64 + f] = x[(size_t)node * 64 + f] * sc;
    newpos[node] = pos;
  } else {
    newpos[node] = -1;
  }
}

__global__ void remap_edges(const int* __restrict__ newpos, int* srcb, int* dstb, int* maskb) {
  int e = blockIdx.x * blockDim.x + threadIdx.x;
  if (e >= E_) return;
  int m = maskb[e];
  int s2 = newpos[srcb[e]], d2 = newpos[dstb[e]];
  m = m && (s2 >= 0) && (d2 >= 0);
  srcb[e] = m ? s2 : 0;
  dstb[e] = m ? d2 : 0;
  maskb[e] = m;
}

// ---- readout: rep[g] += [max over nodes || mean over nodes] ----------------
__global__ void readout(const float* __restrict__ x, int kk, float* __restrict__ rep) {
  int g = blockIdx.x, f = threadIdx.x;  // 64 threads
  float mx = -1e30f, sm = 0.f;
  for (int i = 0; i < kk; ++i) {
    float vv = x[(size_t)(g * kk + i) * 64 + f];
    mx = fmaxf(mx, vv);
    sm += vv;
  }
  rep[g * 128 + f] += mx;
  rep[g * 128 + 64 + f] += sm / (float)kk;
}

// ---- head: sigmoid(h3 @ W4 + b4) -------------------------------------------
__global__ void final_head(const float* __restrict__ h3, const float* __restrict__ W4,
                           const float* __restrict__ b4, float* __restrict__ out) {
  int g = threadIdx.x;  // 256 threads, 1 block
  float s = b4[0];
  for (int f = 0; f < 64; ++f) s += h3[g * 64 + f] * W4[f];
  out[g] = 1.f / (1.f + expf(-s));
}

extern "C" void kernel_launch(void* const* d_in, const int* in_sizes, int n_in,
                              void* d_out, int out_size, void* d_ws, size_t ws_size,
                              hipStream_t stream) {
  (void)in_sizes; (void)n_in; (void)out_size; (void)ws_size;
  // ---- inputs (setup_inputs insertion order, params pytree in insertion order)
  const float* dx = (const float*)d_in[0];          // [NT,128]
  const float* dew = (const float*)d_in[1];         // [E,16]
  const int*   dei = (const int*)d_in[2];           // [2,E]
  int p = 4;                                        // d_in[3] = batch (unused)
  const float* cv[4][10];                           // Wq,bq,Wk,bk,Wv,bv,We,Ws,bs,Wb
  for (int l = 0; l < 4; ++l)
    for (int j = 0; j < 10; ++j) cv[l][j] = (const float*)d_in[p++];
  const float *Wt[4], *bt[4], *bng[4], *bnb[4], *poolw[3], *mlpW[4], *mlpb[4];
  for (int l = 0; l < 4; ++l) { Wt[l] = (const float*)d_in[p++]; bt[l] = (const float*)d_in[p++]; }
  for (int l = 0; l < 4; ++l) { bng[l] = (const float*)d_in[p++]; bnb[l] = (const float*)d_in[p++]; }
  for (int l = 0; l < 3; ++l) poolw[l] = (const float*)d_in[p++];
  for (int l = 0; l < 4; ++l) { mlpW[l] = (const float*)d_in[p++]; mlpb[l] = (const float*)d_in[p++]; }

  // ---- workspace carve-up ----
  float* ws = (float*)d_ws;
  size_t o = 0;
  auto alloc = [&](size_t n) { float* r = ws + o; o += n; return r; };
  float* q    = alloc((size_t)NT_ * HC_);
  float* kbuf = alloc((size_t)NT_ * HC_);
  float* v    = alloc((size_t)NT_ * HC_);
  float* xr   = alloc((size_t)NT_ * HC_);
  float* outa = alloc((size_t)NT_ * HC_);
  float* tc   = alloc((size_t)NT_ * HC_);
  float* alpha = alloc((size_t)E_ * H_);
  float* den   = alloc((size_t)NT_ * H_);
  unsigned* amaxb = (unsigned*)alloc((size_t)NT_ * H_);
  float* xbuf0 = alloc((size_t)NT_ * 64);
  float* xbuf1 = alloc((size_t)NT_ * 64);
  float* rep = alloc((size_t)B_ * 128);
  float* h1 = alloc((size_t)B_ * 256);
  float* h2 = alloc((size_t)B_ * 128);
  float* h3 = alloc((size_t)B_ * 64);
  _Float16* a16 = (_Float16*)alloc((size_t)NT_ * HC_ / 2 + 64);  // M*K halfs (max 32768*192)
  _Float16* wp  = (_Float16*)alloc(40000);                       // packed weights (max 32K halfs)
  int* newpos = (int*)alloc(NT_);
  int* srcb = (int*)alloc(E_);
  int* dstb = (int*)alloc(E_);
  int* maskb = (int*)alloc(E_);

  float neg = -1e9f; unsigned nb; memcpy(&nb, &neg, 4);
  const unsigned initkey = ~nb;   // encoded -1e9 (negative float)

  auto cvtA = [&](const float* A, int n) {
    cvt_f16<<<CDIV(CDIV(n, 4), 256), 256, 0, stream>>>(A, a16, n);
  };
  auto gemm = [&](const float* W, const float* bias, float* Co,
                  int M, int N, int K, int act, const float* g, const float* bb) {
    pack_b<<<CDIV((N / 16) * (K / 32) * 32, 256), 256, 0, stream>>>(W, wp, K, N);
    gemm16<<<dim3(M / 64, N / 16), 32, 0, stream>>>(a16, wp, bias, Co, M, N, K, act, g, bb);
  };

  copy_edges<<<CDIV(E_, 256), 256, 0, stream>>>(dei, srcb, dstb, maskb);
  fill_f32<<<CDIV(B_ * 128, 256), 256, 0, stream>>>(rep, 0.f, B_ * 128);

  const int Ns[4] = {NT_, NT_, NT_ / 2, NT_ / 4};   // conv input node counts
  const int nper[4] = {0, 128, 64, 32};             // per-graph size pre-pool

  for (int l = 0; l < 4; ++l) {
    const int Ncur = Ns[l];
    const int din = (l == 0) ? FIN_ : 64;
    const float* xin = (l == 0) ? dx : ((l == 1) ? xbuf0 : xbuf1);
    // projections (WMMA GEMMs; activations converted to f16 once per layer)
    cvtA(xin, Ncur * din);
    gemm(cv[l][0], cv[l][1], q,    Ncur, HC_, din, 0, nullptr, nullptr);
    gemm(cv[l][2], cv[l][3], kbuf, Ncur, HC_, din, 0, nullptr, nullptr);
    gemm(cv[l][4], cv[l][5], v,    Ncur, HC_, din, 0, nullptr, nullptr);
    gemm(cv[l][7], cv[l][8], xr,   Ncur, HC_, din, 0, nullptr, nullptr);
    // segment softmax state
    fill_u32<<<CDIV(Ncur * H_, 256), 256, 0, stream>>>(amaxb, initkey, Ncur * H_);
    fill_f32<<<CDIV(Ncur * H_, 256), 256, 0, stream>>>(den, 0.f, Ncur * H_);
    fill_f32<<<CDIV(Ncur * HC_, 256), 256, 0, stream>>>(outa, 0.f, Ncur * HC_);
    // edge passes
    attn_alpha<<<CDIV(E_ * H_, 256), 256, 0, stream>>>(q, kbuf, dew, cv[l][6],
                                                       srcb, dstb, maskb, alpha, amaxb);
    attn_exden<<<CDIV(E_ * H_, 256), 256, 0, stream>>>(alpha, amaxb, dstb, maskb, den);
    attn_msg<<<CDIV(E_ * H_, 256), 256, 0, stream>>>(alpha, den, v, dew, cv[l][6],
                                                     srcb, dstb, maskb, outa);
    beta_combine<<<CDIV(Ncur, 256), 256, 0, stream>>>(outa, xr, cv[l][9], tc, Ncur);
    // transf + ReLU + BN (fused WMMA GEMM epilogue)
    cvtA(tc, Ncur * HC_);
    gemm(Wt[l], bt[l], xbuf0, Ncur, 64, HC_, 1, bng[l], bnb[l]);
    if (l >= 1) {
      int n = nper[l], kk = n / 2;
      topk_pool<<<B_, n, 0, stream>>>(xbuf0, poolw[l - 1], n, kk, xbuf1, newpos);
      remap_edges<<<CDIV(E_, 256), 256, 0, stream>>>(newpos, srcb, dstb, maskb);
      readout<<<B_, 64, 0, stream>>>(xbuf1, kk, rep);
    }
  }

  // MLP head (WMMA GEMMs) + sigmoid
  cvtA(rep, B_ * 128);
  gemm(mlpW[0], mlpb[0], h1, B_, 256, 128, 1, nullptr, nullptr);
  cvtA(h1, B_ * 256);
  gemm(mlpW[1], mlpb[1], h2, B_, 128, 256, 1, nullptr, nullptr);
  cvtA(h2, B_ * 128);
  gemm(mlpW[2], mlpb[2], h3, B_, 64, 128, 0, nullptr, nullptr);
  final_head<<<1, B_, 0, stream>>>(h3, mlpW[3], mlpb[3], (float*)d_out);
}
